// Attention_18648747999422
// MI455X (gfx1250) — compile-verified, matrix operand order checked by reading, not run on hardware
//
#include <hip/hip_runtime.h>
#include <hip/hip_bf16.h>
#include <cstddef>

// Problem constants (match reference)
#define B_   32
#define S_   2048
#define ENC_ 1024
#define DEC_ 1024

// LDS tile geometry for the energy kernel
#define ROWS_PER_BLK 128            // 8 waves * 16 rows
#define STR (1024 + 8)              // bf16 row stride (pad 8 shorts) -> conflict-free b128 reads
#define ENC_SHORTS (ROWS_PER_BLK * STR)
#define WE_SHORTS  (16 * STR)
// enc tile + w_e tile + hproj row + v_w  (bytes)
#define SMEM_BYTES ((ENC_SHORTS + WE_SHORTS) * 2 + 2 * DEC_ * 4)   // 305,408 B < 320KB

typedef __attribute__((ext_vector_type(16))) __bf16 v16bf;
typedef __attribute__((ext_vector_type(8)))  float  v8f;

union alignas(32) Frag { unsigned int u[8]; uint4 q[2]; v16bf v; };

// Hardware tanh if the toolchain exposes it (V_TANH_F32 on gfx1250);
// falls back to libm tanhf so compilation can never break.
#if defined(__has_builtin)
#if __has_builtin(__builtin_amdgcn_tanhf)
#define FAST_TANH(x) __builtin_amdgcn_tanhf(x)
#endif
#endif
#ifndef FAST_TANH
#define FAST_TANH(x) tanhf(x)
#endif

// Packed f32 -> bf16x2 convert: single V_CVT_PK_BF16_F32 when available.
#if defined(__has_builtin)
#if __has_builtin(__builtin_amdgcn_cvt_pk_bf16_f32)
__device__ __forceinline__ unsigned int pack_bf16(float a, float b) {
    return __builtin_bit_cast(unsigned int, __builtin_amdgcn_cvt_pk_bf16_f32(a, b));
}
#define HAVE_PK_BF16 1
#endif
#endif
#ifndef HAVE_PK_BF16
__device__ __forceinline__ unsigned int pack_bf16(float a, float b) {
    unsigned int ua = __float_as_uint(a), ub = __float_as_uint(b);
    ua = (ua + 0x7FFFu + ((ua >> 16) & 1u)) >> 16;      // round-to-nearest-even
    ub = (ub + 0x7FFFu + ((ub >> 16) & 1u)) >> 16;
    return (ua & 0xFFFFu) | (ub << 16);
}
#endif

// ---------------------------------------------------------------------------
// Kernel 1: h_proj[b,k] = hidden[b,:] . attn_w[k, 0:DEC] + attn_b[k]
// One wave per k (coalesced along d), grid = (DEC/8, B), block = 256 (8 waves)
// ---------------------------------------------------------------------------
__global__ __launch_bounds__(256) void hproj_kernel(
    const float* __restrict__ hidden, const float* __restrict__ attn_w,
    const float* __restrict__ attn_b, float* __restrict__ hproj)
{
    const int b    = blockIdx.y;
    const int lane = threadIdx.x & 31;
    const int wave = threadIdx.x >> 5;
    const int k    = blockIdx.x * 8 + wave;
    float acc = 0.f;
    for (int d = lane; d < DEC_; d += 32)
        acc += hidden[(size_t)b * DEC_ + d] * attn_w[(size_t)k * (ENC_ + DEC_) + d];
    acc += __shfl_xor(acc, 1, 32);
    acc += __shfl_xor(acc, 2, 32);
    acc += __shfl_xor(acc, 4, 32);
    acc += __shfl_xor(acc, 8, 32);
    acc += __shfl_xor(acc, 16, 32);
    if (lane == 0) hproj[(size_t)b * DEC_ + k] = acc + attn_b[k];
}

// ---------------------------------------------------------------------------
// Kernel 2: WMMA bf16 GEMM + fused tanh / v_w epilogue -> scores[b,s]
// grid = (S/128, B), block = 256 (8 waves, one 16-row M tile per wave)
// LDS: encT[128][STR] bf16 (staged once) + weT[16][STR] bf16 (per N tile)
//      + hproj row + v_w (f32)
// Each operand fragment = 2 x ds_load_b128; dual accumulator chains over
// the split K range give independent WMMA streams per wave. The next w_e
// tile is prefetched (global_prefetch_b8) while the current one computes.
// ---------------------------------------------------------------------------
__global__ __launch_bounds__(256) void energy_kernel(
    const float* __restrict__ enc, const float* __restrict__ attn_w,
    const float* __restrict__ hproj, const float* __restrict__ vw,
    float* __restrict__ scores)
{
    extern __shared__ unsigned short smem_dyn[];
    unsigned short* encT = smem_dyn;                 // 128 x STR bf16
    unsigned short* weT  = smem_dyn + ENC_SHORTS;    // 16  x STR bf16
    float* hpS = (float*)(smem_dyn + ENC_SHORTS + WE_SHORTS);   // DEC_ f32
    float* vwS = hpS + DEC_;                                    // DEC_ f32

    const int tid   = threadIdx.x;
    const int b     = blockIdx.y;
    const int sBase = blockIdx.x * ROWS_PER_BLK;

    // --- stage hproj row (bias already fused) + v_w ---
    #pragma unroll
    for (int i = 0; i < DEC_ / 256; ++i) {
        hpS[tid + i * 256] = hproj[(size_t)b * DEC_ + tid + i * 256];
        vwS[tid + i * 256] = vw[tid + i * 256];
    }

    // --- stage encoder tile (f32 -> bf16), 32768 quads / 256 threads ---
    for (int i = 0; i < 128; ++i) {
        int p = tid + i * 256;
        int r = p >> 8;                 // 256 quads per row
        int e = (p & 255) << 2;
        const float4 f = *(const float4*)(enc + ((size_t)b * S_ + sBase + r) * ENC_ + e);
        uint2 v; v.x = pack_bf16(f.x, f.y); v.y = pack_bf16(f.z, f.w);
        *(uint2*)(encT + (size_t)r * STR + e) = v;
    }

    const int lane = tid & 31;
    const int wave = tid >> 5;
    const int half = lane >> 4;
    const int l15  = lane & 15;
    const int Mrow = wave * 16 + l15;   // A-fragment row for this lane

    const unsigned short* aBase = encT + (size_t)Mrow * STR + 8 * half;
    const unsigned short* bBase = weT  + (size_t)l15  * STR + 8 * half;

    float part[8] = {0.f, 0.f, 0.f, 0.f, 0.f, 0.f, 0.f, 0.f};

    for (int nt = 0; nt < DEC_ / 16; ++nt) {
        __syncthreads();                // previous tile fully consumed
        // --- stage w_e tile rows k = nt*16 .. +15 (f32 -> bf16) ---
        #pragma unroll
        for (int i = 0; i < 16; ++i) {
            int p  = tid + i * 256;
            int rr = p >> 8;
            int e  = (p & 255) << 2;
            const float4 f = *(const float4*)(attn_w + (size_t)(nt * 16 + rr) * (ENC_ + DEC_) + DEC_ + e);
            uint2 v; v.x = pack_bf16(f.x, f.y); v.y = pack_bf16(f.z, f.w);
            *(uint2*)(weT + (size_t)rr * STR + e) = v;
        }
        __syncthreads();

        // prefetch next w_e tile (16 rows x 4KB, row pitch 8KB) into cache
        // while this tile computes; global_prefetch_b8 has no counter cost.
        if (nt + 1 < DEC_ / 16) {
            const char* nx = (const char*)(attn_w + (size_t)((nt + 1) * 16) * (ENC_ + DEC_) + DEC_);
            #pragma unroll
            for (int i = 0; i < 4; ++i) {
                int line = tid + i * 256;          // 1024 x 64B lines
                int rr   = line >> 6;
                int off  = (line & 63) << 6;
                __builtin_prefetch(nx + (size_t)rr * (ENC_ + DEC_) * 4 + off, 0, 2);
            }
        }

        // dual accumulator chains over split K range (e: 0..511 and 512..1023)
        v8f acc0 = {0.f, 0.f, 0.f, 0.f, 0.f, 0.f, 0.f, 0.f};
        v8f acc1 = {0.f, 0.f, 0.f, 0.f, 0.f, 0.f, 0.f, 0.f};
        #pragma unroll 4
        for (int e0 = 0; e0 < ENC_ / 2; e0 += 32) {
            Frag a0, b0, a1, b1;
            a0.q[0] = *(const uint4*)(aBase + e0);
            a0.q[1] = *(const uint4*)(aBase + e0 + 16);
            b0.q[0] = *(const uint4*)(bBase + e0);
            b0.q[1] = *(const uint4*)(bBase + e0 + 16);
            a1.q[0] = *(const uint4*)(aBase + e0 + ENC_ / 2);
            a1.q[1] = *(const uint4*)(aBase + e0 + ENC_ / 2 + 16);
            b1.q[0] = *(const uint4*)(bBase + e0 + ENC_ / 2);
            b1.q[1] = *(const uint4*)(bBase + e0 + ENC_ / 2 + 16);
            acc0 = __builtin_amdgcn_wmma_f32_16x16x32_bf16(
                false, a0.v, false, b0.v, (short)0, acc0, false, false);
            acc1 = __builtin_amdgcn_wmma_f32_16x16x32_bf16(
                false, a1.v, false, b1.v, (short)0, acc1, false, false);
        }
        v8f acc = acc0 + acc1;

        // fused epilogue: tanh(e_proj + h_proj + bias) * v_w, per (row, n) cell
        const int   n  = nt * 16 + l15;
        const float hp = hpS[n];
        const float vv = vwS[n];
        #pragma unroll
        for (int r = 0; r < 8; ++r) {
            float en = FAST_TANH(acc[r] + hp);
            part[r] += en * vv;
        }
    }

    // reduce over the 16 lanes of each half (N columns), then store per-row score
    #pragma unroll
    for (int r = 0; r < 8; ++r) {
        part[r] += __shfl_xor(part[r], 1, 32);
        part[r] += __shfl_xor(part[r], 2, 32);
        part[r] += __shfl_xor(part[r], 4, 32);
        part[r] += __shfl_xor(part[r], 8, 32);
    }
    #pragma unroll
    for (int r = 0; r < 8; ++r) {
        if (l15 == r) {   // lane r -> row r (half 0), lane 16+r -> row 8+r (half 1)
            int row = sBase + wave * 16 + r + 8 * half;
            scores[(size_t)b * S_ + row] = part[r];
        }
    }
}

// ---------------------------------------------------------------------------
// Kernel 3: masked softmax over S per batch; writes weights to d_out and ws
// ---------------------------------------------------------------------------
__global__ __launch_bounds__(256) void softmax_kernel(
    float* __restrict__ scores, const int* __restrict__ mask,
    float* __restrict__ attn_out)
{
    __shared__ float red[8];
    const int b = blockIdx.x, tid = threadIdx.x;
    const int lane = tid & 31, wave = tid >> 5;

    float vals[8];
    float mx = -3.4e38f;
    #pragma unroll
    for (int i = 0; i < 8; ++i) {
        int s = tid + i * 256;
        float sc = scores[(size_t)b * S_ + s];
        sc = mask[(size_t)b * S_ + s] ? sc : -1.0e10f;
        vals[i] = sc;
        mx = fmaxf(mx, sc);
    }
    mx = fmaxf(mx, __shfl_xor(mx, 1, 32));
    mx = fmaxf(mx, __shfl_xor(mx, 2, 32));
    mx = fmaxf(mx, __shfl_xor(mx, 4, 32));
    mx = fmaxf(mx, __shfl_xor(mx, 8, 32));
    mx = fmaxf(mx, __shfl_xor(mx, 16, 32));
    if (lane == 0) red[wave] = mx;
    __syncthreads();
    mx = red[0];
    #pragma unroll
    for (int j = 1; j < 8; ++j) mx = fmaxf(mx, red[j]);
    __syncthreads();

    float sum = 0.f;
    #pragma unroll
    for (int i = 0; i < 8; ++i) { vals[i] = __expf(vals[i] - mx); sum += vals[i]; }
    sum += __shfl_xor(sum, 1, 32);
    sum += __shfl_xor(sum, 2, 32);
    sum += __shfl_xor(sum, 4, 32);
    sum += __shfl_xor(sum, 8, 32);
    sum += __shfl_xor(sum, 16, 32);
    if (lane == 0) red[wave] = sum;
    __syncthreads();
    sum = 0.f;
    #pragma unroll
    for (int j = 0; j < 8; ++j) sum += red[j];
    const float inv = 1.0f / sum;

    #pragma unroll
    for (int i = 0; i < 8; ++i) {
        int s = tid + i * 256;
        float w = vals[i] * inv;
        scores[(size_t)b * S_ + s]   = w;   // kept for context kernel
        attn_out[(size_t)b * S_ + s] = w;   // output #2
    }
}

// ---------------------------------------------------------------------------
// Kernel 4: context[b,e] = sum_s w[b,s] * enc[b,s,e]  (coalesced over e)
// grid = (ENC/256, B), block = 256
// ---------------------------------------------------------------------------
__global__ __launch_bounds__(256) void context_kernel(
    const float* __restrict__ wts, const float* __restrict__ enc,
    float* __restrict__ ctx)
{
    __shared__ float w[S_];
    const int b = blockIdx.y, tid = threadIdx.x;
    #pragma unroll
    for (int i = 0; i < S_ / 256; ++i)
        w[tid + i * 256] = wts[(size_t)b * S_ + tid + i * 256];
    __syncthreads();
    const int e = blockIdx.x * 256 + tid;
    float acc = 0.f;
    for (int s = 0; s < S_; ++s)
        acc += w[s] * enc[((size_t)b * S_ + s) * ENC_ + e];
    ctx[(size_t)b * ENC_ + e] = acc;
}

// ---------------------------------------------------------------------------
extern "C" void kernel_launch(void* const* d_in, const int* in_sizes, int n_in,
                              void* d_out, int out_size, void* d_ws, size_t ws_size,
                              hipStream_t stream)
{
    const float* hidden = (const float*)d_in[0];
    const float* enc    = (const float*)d_in[1];
    const int*   mask   = (const int*)  d_in[2];
    const float* attn_w = (const float*)d_in[3];
    const float* attn_b = (const float*)d_in[4];
    const float* v_w    = (const float*)d_in[5];

    float* ctx_out  = (float*)d_out;                 // B*ENC floats
    float* attn_out = (float*)d_out + B_ * ENC_;     // B*S floats

    float* hproj  = (float*)d_ws;                                        // B*DEC f32
    float* scores = (float*)((char*)d_ws + (size_t)B_ * DEC_ * sizeof(float)); // B*S f32

    hproj_kernel<<<dim3(DEC_ / 8, B_), 256, 0, stream>>>(hidden, attn_w, attn_b, hproj);

    // Opt into >64KB dynamic LDS (WGP has 320KB on CDNA5). Host-side attr set
    // is deterministic and capture-safe (not a stream op).
    (void)hipFuncSetAttribute((const void*)energy_kernel,
                              hipFuncAttributeMaxDynamicSharedMemorySize, SMEM_BYTES);
    energy_kernel<<<dim3(S_ / ROWS_PER_BLK, B_), 256, SMEM_BYTES, stream>>>(
        enc, attn_w, hproj, v_w, scores);

    softmax_kernel<<<B_, 256, 0, stream>>>(scores, mask, attn_out);

    context_kernel<<<dim3(ENC_ / 256, B_), 256, 0, stream>>>(scores, enc, ctx_out);
}
